// BinaryTreeGoE_26525718020584
// MI455X (gfx1250) — matrix operand order repaired and checked by reading, not compiled, not run on hardware
//
#include <hip/hip_runtime.h>

// ---------------------------------------------------------------------------
// BinaryTreeGoE on MI455X (gfx1250): expert-binned bf16 WMMA GEMM per level,
// with Tensor-Data-Mover (TDM) double-buffered LDS staging of the B matrix.
//   B=4096 samples, D=2048, DEPTH=4, E=2^d experts at level d.
// Block = 256 threads = 8 waves (4 row-groups x 2 col-groups).
// Block tile 64 rows x 256 cols; wave tile 16 x 128 (8 wmma accumulators).
// ---------------------------------------------------------------------------

#define BB 4096
#define DD 2048
#define NCHUNK (DD / 32)        // 64 K-chunks of 32
#define LDS_PITCH 72            // 64B of k-data + 8B TDM pad per n-row
#define CHUNK_BYTES (256 * LDS_PITCH)

typedef __attribute__((ext_vector_type(4)))  __bf16 v4bf;
typedef __attribute__((ext_vector_type(8)))  __bf16 v8bf;
typedef __attribute__((ext_vector_type(16))) __bf16 v16bf;
typedef __attribute__((ext_vector_type(8)))  float  v8f;
typedef __attribute__((ext_vector_type(4)))  unsigned int v4u;
typedef __attribute__((ext_vector_type(8)))  int v8i;
typedef __attribute__((ext_vector_type(4)))  int v4i;

#if __has_builtin(__builtin_amdgcn_tensor_load_to_lds)
#define HAVE_TDM 1
#else
#define HAVE_TDM 0
#endif

// The TDM builtin carries no pointer argument, so alias analysis cannot
// connect its LDS writes to ldsB.  Passing the LDS pointer INTO a volatile
// asm makes the object escape (captured), and the "memory" clobber then
// forces the compiler to treat it as written by the DMA — so the ds_load
// fragment reads are real and ordered after s_wait_tensorcnt + barrier.
static __device__ inline void lds_written_fence(const void* p) {
  asm volatile("" : : "v"(p) : "memory");
}

// A fragment (16-bit A 16x32): lane<16 -> K k..k+15;
// lane>=16 -> K k+16..k+31.  p already includes the +16 half shift.
static __device__ inline v16bf ld_frag_a(const __bf16* p) {
  v8bf lo = *(const v8bf*)(p);
  v8bf hi = *(const v8bf*)(p + 8);
  return __builtin_shufflevector(lo, hi, 0,1,2,3,4,5,6,7,8,9,10,11,12,13,14,15);
}

// B fragment (16-bit B 32x16): lane<16 -> K k..k+15 @ N=lane; lane>=16 ->
// K k+16..k+31.  Reads 32B from an LDS row (pitch 72B -> only 8B aligned),
// so use 4x 8B loads (ds_load_b64 / 2addr pairs).
static __device__ inline v16bf ld_frag_b_lds(const char* p) {
  v4bf a0 = *(const v4bf*)(p);
  v4bf a1 = *(const v4bf*)(p + 8);
  v4bf a2 = *(const v4bf*)(p + 16);
  v4bf a3 = *(const v4bf*)(p + 24);
  v8bf lo = __builtin_shufflevector(a0, a1, 0,1,2,3,4,5,6,7);
  v8bf hi = __builtin_shufflevector(a2, a3, 0,1,2,3,4,5,6,7);
  return __builtin_shufflevector(lo, hi, 0,1,2,3,4,5,6,7,8,9,10,11,12,13,14,15);
}

#if HAVE_TDM
// One TDM descriptor: 2-D tile, x = 32 k-elements (64B row), y = 256 n-rows,
// element 2B, row stride = DD elements, LDS pad 8B every 64B (pitch 72B).
static __device__ inline void tdm_load_b_chunk(const __bf16* gsrc, unsigned lds_off) {
  unsigned long long ga = (unsigned long long)(uintptr_t)gsrc;
  v4u g0;
  g0[0] = 1u;                                              // count=1 (valid, user)
  g0[1] = lds_off;                                         // lds_addr (bytes)
  g0[2] = (unsigned)(ga & 0xFFFFFFFFu);                    // global_addr[31:0]
  g0[3] = (unsigned)((ga >> 32) & 0x1FFFFFFu) | (2u << 30);// addr[56:32] | type=2
  v8i g1;
  g1[0] = (int)((1u << 16) |        // data_size = 1 -> 2 bytes
                (1u << 20) |        // pad_enable
                (3u << 22) |        // pad_interval code 3 -> every 64B
                (1u << 25));        // pad_amount  code 1 -> 8B
  g1[1] = (int)((unsigned)DD << 16);   // tensor_dim0 low 16 into bits[31:16]
  g1[2] = (int)((unsigned)DD << 16);   // tensor_dim0 hi=0 | tensor_dim1 low=DD
  g1[3] = (int)(32u << 16);            // tensor_dim1 hi=0 | tile_dim0=32
  g1[4] = 256;                          // tile_dim1=256, tile_dim2=0
  g1[5] = DD;                           // tensor_dim0_stride low 32
  g1[6] = 0;                            // stride hi | tensor_dim1_stride low
  g1[7] = 0;
  v4i g2; g2[0] = 0; g2[1] = 0; g2[2] = 0; g2[3] = 0;
  v4i g3; g3[0] = 0; g3[1] = 0; g3[2] = 0; g3[3] = 0;
  v8i g4; g4[0] = 0; g4[1] = 0; g4[2] = 0; g4[3] = 0;
          g4[4] = 0; g4[5] = 0; g4[6] = 0; g4[7] = 0;
  __builtin_amdgcn_tensor_load_to_lds(g0, g1, g2, g3, g4, 0);
}
#endif

// --------------------------- helper kernels --------------------------------

__global__ void cvt_x_kernel(const float* __restrict__ x, __bf16* __restrict__ h) {
  int i = (blockIdx.x * 256 + threadIdx.x) * 4;
  float4 v = *(const float4*)(x + i);
  v4bf o;
  o.x = (__bf16)v.x; o.y = (__bf16)v.y; o.z = (__bf16)v.z; o.w = (__bf16)v.w;
  *(v4bf*)(h + i) = o;
}

// Wt[e][n][k] = bf16(W[e][k][n]); tiled via LDS, coalesced both sides.
__global__ void transpose_cvt_kernel(const float* __restrict__ W, __bf16* __restrict__ Wt) {
  __shared__ float t[32][33];
  const int e  = blockIdx.z;
  const int n0 = blockIdx.x * 32;
  const int k0 = blockIdx.y * 32;
  const float* Wp = W + (size_t)e * DD * DD;
  __bf16*      Wo = Wt + (size_t)e * DD * DD;
#pragma unroll
  for (int j = 0; j < 32; j += 8)
    t[threadIdx.y + j][threadIdx.x] =
        Wp[(size_t)(k0 + threadIdx.y + j) * DD + n0 + threadIdx.x];
  __syncthreads();
#pragma unroll
  for (int j = 0; j < 32; j += 8)
    Wo[(size_t)(n0 + threadIdx.y + j) * DD + k0 + threadIdx.x] =
        (__bf16)t[threadIdx.x][threadIdx.y + j];
}

__global__ void zero_counts_kernel(int* counts) {
  if (threadIdx.x < 16) counts[threadIdx.x] = 0;
}

__global__ void bin_rows_kernel(const int* __restrict__ pm, int level,
                                int* __restrict__ counts, int* __restrict__ rows) {
  int i = blockIdx.x * 256 + threadIdx.x;
  if (i >= BB) return;
  int e = 0;
  for (int d = 0; d < level; ++d) e = e * 2 + pm[i * 4 + d];
  int pos = atomicAdd(&counts[e], 1);
  rows[e * BB + pos] = i;
}

// --------------------------- main GEMM kernel ------------------------------
// grid = (DD/256, BB/64, E)

__global__ __launch_bounds__(256)
void moe_gemm_kernel(const __bf16* __restrict__ hin,
                     const __bf16* __restrict__ Wt,
                     const float*  __restrict__ bias,
                     const int*    __restrict__ counts,
                     const int*    __restrict__ rows,
                     __bf16* __restrict__ hout_bf,
                     float*  __restrict__ hout_f32,
                     int write_f32) {
  __shared__ __align__(16) char ldsB[2][CHUNK_BYTES];   // 2 x 18 KB

  const int e   = blockIdx.z;
  const int cnt = counts[e];
  const int m_base = blockIdx.y * 64;
  if (m_base >= cnt) return;

  const int tid  = threadIdx.x;
  const int wave = tid >> 5;
  const int lane = tid & 31;
  const int wrow = wave >> 1;      // 0..3
  const int wcol = wave & 1;       // 0..1
  const int lm   = lane & 15;
  const int hi   = lane >> 4;      // 0 or 1

  const int m0 = m_base + wrow * 16;
  const int n0 = blockIdx.x * 256 + wcol * 128;

  const int* rlist = rows + e * BB;
  const int midx   = m0 + lm;
  const int arow   = rlist[midx < cnt ? midx : cnt - 1];

  const __bf16* aptr  = hin + (size_t)arow * DD + (hi << 4);   // +16 elems upper half
  const __bf16* bsrc  = Wt + (size_t)e * DD * DD + (size_t)(blockIdx.x * 256) * DD;

  // Per-lane LDS read base for this wave's B fragments: local row = wave cols.
  const int lrow0 = wcol * 128 + lm;                      // + t*16 per tile
  const unsigned lbase = (unsigned)(lrow0 * LDS_PITCH + hi * 32);

  v8f acc[8];
#pragma unroll
  for (int t = 0; t < 8; ++t) acc[t] = (v8f)(0.0f);

#if HAVE_TDM
  // ---- prologue: DMA chunk 0 into buffer 0 ----
  if (wave == 0) {
    tdm_load_b_chunk(bsrc, 0u);                 // ldsB assumed at LDS offset 0
    __builtin_amdgcn_s_wait_tensorcnt(0);
  }
  __syncthreads();
  lds_written_fence(&ldsB[0][0]);

  v16bf aCur = ld_frag_a(aptr);
  for (int c = 0; c < NCHUNK; ++c) {
    const int nxt = c + 1;
    if (wave == 0 && nxt < NCHUNK)              // overlap DMA(c+1) with compute(c)
      tdm_load_b_chunk(bsrc + nxt * 32, (unsigned)((nxt & 1) * CHUNK_BYTES));

    v16bf aNext = ld_frag_a(aptr + (nxt < NCHUNK ? nxt * 32 : 0));

    // B fragments pipelined one stage: ds loads for t+1 overlap wmma t, and
    // their address VALU writes fresh regs (no WMMA->VALU WAR hazard nops).
    const char* lb = &ldsB[c & 1][lbase];
    v16bf bCur = ld_frag_b_lds(lb);
#pragma unroll
    for (int t = 0; t < 8; ++t) {
      v16bf bNext = (t < 7) ? ld_frag_b_lds(lb + (t + 1) * 16 * LDS_PITCH) : bCur;
      acc[t] = __builtin_amdgcn_wmma_f32_16x16x32_bf16(
          false, aCur, false, bCur, (short)0, acc[t], false, false);
      bCur = bNext;
    }

    if (wave == 0) __builtin_amdgcn_s_wait_tensorcnt(0);
    __syncthreads();
    lds_written_fence(&ldsB[0][0]);
    aCur = aNext;
  }
#else
  // ---- fallback: register-staged cooperative copy, double buffered ----
  {
    const int crow = tid >> 2;                  // 64 rows per pass, 4 passes
    const int coff = (tid & 3) * 8;             // elements within 32-elem row
#pragma unroll
    for (int p = 0; p < 4; ++p) {
      int r = crow + p * 64;
      v8bf v = *(const v8bf*)(bsrc + (size_t)r * DD + coff);
      char* dst = &ldsB[0][r * LDS_PITCH + coff * 2];
      *(v4bf*)(dst)     = __builtin_shufflevector(v, v, 0,1,2,3);
      *(v4bf*)(dst + 8) = __builtin_shufflevector(v, v, 4,5,6,7);
    }
  }
  __syncthreads();

  v16bf aCur = ld_frag_a(aptr);
  for (int c = 0; c < NCHUNK; ++c) {
    const int nxt = c + 1;
    const int crow = tid >> 2;
    const int coff = (tid & 3) * 8;
    v8bf stage[4];
    if (nxt < NCHUNK) {
#pragma unroll
      for (int p = 0; p < 4; ++p)
        stage[p] = *(const v8bf*)(bsrc + (size_t)(crow + p * 64) * DD + nxt * 32 + coff);
    }
    v16bf aNext = ld_frag_a(aptr + (nxt < NCHUNK ? nxt * 32 : 0));

    const char* lb = &ldsB[c & 1][lbase];
    v16bf bCur = ld_frag_b_lds(lb);
#pragma unroll
    for (int t = 0; t < 8; ++t) {
      v16bf bNext = (t < 7) ? ld_frag_b_lds(lb + (t + 1) * 16 * LDS_PITCH) : bCur;
      acc[t] = __builtin_amdgcn_wmma_f32_16x16x32_bf16(
          false, aCur, false, bCur, (short)0, acc[t], false, false);
      bCur = bNext;
    }
    __syncthreads();
    if (nxt < NCHUNK) {
#pragma unroll
      for (int p = 0; p < 4; ++p) {
        char* dst = &ldsB[nxt & 1][(crow + p * 64) * LDS_PITCH + coff * 2];
        *(v4bf*)(dst)     = __builtin_shufflevector(stage[p], stage[p], 0,1,2,3);
        *(v4bf*)(dst + 8) = __builtin_shufflevector(stage[p], stage[p], 4,5,6,7);
      }
    }
    __syncthreads();
    aCur = aNext;
  }
#endif

  // Epilogue: C layout (VGPR r: lanes 0-15 -> M=r, lanes 16-31 -> M=r+8).
  const float* bp = bias + (size_t)e * DD;
#pragma unroll
  for (int t = 0; t < 8; ++t) {
    const int n = n0 + t * 16 + lm;
    const float bv = bp[n];
#pragma unroll
    for (int r = 0; r < 8; ++r) {
      const int mi = m0 + r + (hi ? 8 : 0);
      if (mi < cnt) {
        float v = acc[t][r] + bv;
        v = v > 0.0f ? v : 0.0f;
        const int orow = rlist[mi];
        if (write_f32)
          hout_f32[(size_t)orow * DD + n] = v;
        else
          hout_bf[(size_t)orow * DD + n] = (__bf16)v;
      }
    }
  }
}

// ------------------------------- launcher ----------------------------------

extern "C" void kernel_launch(void* const* d_in, const int* in_sizes, int n_in,
                              void* d_out, int out_size, void* d_ws, size_t ws_size,
                              hipStream_t stream) {
  const float* x  = (const float*)d_in[0];
  const int*   pm = (const int*)d_in[1];
  const float* W[4] = {(const float*)d_in[2], (const float*)d_in[4],
                       (const float*)d_in[6], (const float*)d_in[8]};
  const float* bb[4] = {(const float*)d_in[3], (const float*)d_in[5],
                        (const float*)d_in[7], (const float*)d_in[9]};

  char* ws = (char*)d_ws;
  size_t off = 0;
  __bf16* Wt = (__bf16*)(ws + off); off += (size_t)8 * DD * DD * sizeof(__bf16);  // 64 MB
  __bf16* h0 = (__bf16*)(ws + off); off += (size_t)BB * DD * sizeof(__bf16);      // 16 MB
  __bf16* h1 = (__bf16*)(ws + off); off += (size_t)BB * DD * sizeof(__bf16);      // 16 MB
  int* counts = (int*)(ws + off);   off += 256;
  int* rows   = (int*)(ws + off);   off += (size_t)8 * BB * sizeof(int);

  cvt_x_kernel<<<(BB * DD) / (256 * 4), 256, 0, stream>>>(x, h0);

  __bf16* hin  = h0;
  __bf16* hout = h1;
  for (int d = 0; d < 4; ++d) {
    const int E = 1 << d;
    transpose_cvt_kernel<<<dim3(DD / 32, DD / 32, E), dim3(32, 8), 0, stream>>>(W[d], Wt);
    zero_counts_kernel<<<1, 64, 0, stream>>>(counts);
    bin_rows_kernel<<<BB / 256, 256, 0, stream>>>(pm, d, counts, rows);
    moe_gemm_kernel<<<dim3(DD / 256, BB / 64, E), 256, 0, stream>>>(
        hin, Wt, bb[d], counts, rows, hout, (float*)d_out, d == 3 ? 1 : 0);
    __bf16* tmp = hin; hin = hout; hout = tmp;
  }
}
